// M_12283606466779
// MI455X (gfx1250) — matrix operand order repaired and checked by reading, not compiled, not run on hardware
//
#include <hip/hip_runtime.h>

// ---------------------------------------------------------------------------
// Fused softmax(+dropout) + P@V for [B=4,H=16,S=2048,S=2048] x [.,.,S,D=64]
// Single-pass (online/flash-style) softmax: the 1 GiB score tensor streams
// from HBM exactly once; P@V runs on v_wmma_f32_16x16x32_bf16 (f32 accum).
// V tiles (128 keys) are staged in LDS as transposed bf16, shared by all
// 16 waves of the workgroup; V itself lives in L2 (32 MiB << 192 MB).
// ---------------------------------------------------------------------------

typedef __attribute__((ext_vector_type(16))) __bf16    v16bf;
typedef __attribute__((ext_vector_type(8)))  float     v8f;
typedef __attribute__((ext_vector_type(8)))  unsigned  v8u;

#define S_DIM   2048
#define D_DIM   64
#define KTILE   128                // keys staged in LDS per barrier pair
#define KCHUNK  32                 // keys per WMMA pass
#define WAVES   16
#define NTHREADS (WAVES * 32)      // 512
#define BLOCK_Q (WAVES * 16)       // 256 query rows per workgroup
#define VT_PAD  136                // bf16 per VT row: 272B stride, 16B aligned

#define KEEP_BYTE 230u             // keep if byte < 230  (p ~= 230/256 ~ 0.9)
#define KEEP_P    0.8984375f       // 230/256, used to rebuild the true row sum
#define LOG2E     1.44269504088896340736f
#define LOG2_INV_KEEP 0.15432126310734963f   // log2(256/230)

// pack two floats into two bf16 (low = a, high = b); round-nearest-ties-away
__device__ __forceinline__ unsigned pack_bf16(float a, float b) {
    unsigned ua = __float_as_uint(a) + 0x8000u;
    unsigned ub = __float_as_uint(b) + 0x8000u;
    return __builtin_amdgcn_perm(ub, ua, 0x07060302u);  // {ub.hi16, ua.hi16}
}

// 4 dropout decisions per hash (one byte each)
__device__ __forceinline__ unsigned hash4(unsigned x) {
    x *= 0x9E3779B1u;
    x ^= x >> 16;
    x *= 0x85EBCA77u;
    x ^= x >> 13;
    return x;
}

__global__ __launch_bounds__(NTHREADS)
void fused_softmax_dropout_pv(const float* __restrict__ scores,
                              const float* __restrict__ vmat,
                              float* __restrict__ out)
{
    // Transposed bf16 V tile: VT[d][k], d in [0,64), k in [0,KTILE)
    __shared__ __align__(16) unsigned short VT[D_DIM * VT_PAD];
    unsigned* VT32 = (unsigned*)VT;                 // u32 view, row stride 68

    const int tid  = threadIdx.x;
    const int lane = tid & 31;
    const int wave = tid >> 5;
    const int bh   = blockIdx.y;                    // b*16 + h
    const int qT   = blockIdx.x;                    // 256-row query tile

    const int qRow = qT * BLOCK_Q + wave * 16 + (lane & 15);    // A-space row
    const unsigned rowIdx  = (unsigned)bh * S_DIM + (unsigned)qRow;
    const unsigned rowBase = rowIdx * (unsigned)S_DIM;          // < 2^28
    const float* __restrict__ srow = scores + (size_t)rowIdx * S_DIM;

    // A fragment K-grouping (16-bit A 16x32 ISA layout):
    //   lanes 0-15 : K in {0..7, 16..23};  lanes 16-31 : K in {8..15, 24..31}
    const int g0   = (lane < 16) ? 0 : 8;
    const int half = (lane & 16) ? 8 : 0;           // C-space row offset
    const int nB   = lane & 15;                     // B/C tile column
    const int kB   = (lane < 16) ? 0 : 8;           // B K-offset within chunk

    float m2 = -1.0e30f;                            // running row max * log2(e)
    float l  = 0.0f;                                // running row sum (pre-drop)
    v8f acc[4];
    #pragma unroll
    for (int t = 0; t < 4; ++t)
        #pragma unroll
        for (int i = 0; i < 8; ++i) acc[t][i] = 0.0f;

    const size_t vBase = (size_t)bh * S_DIM * D_DIM;

    for (int tk = 0; tk < S_DIM / KTILE; ++tk) {
        // ---- stage V tile (KTILE x 64 f32 -> transposed bf16 in LDS) -------
        __syncthreads();
        #pragma unroll
        for (int c = 0; c < 2; ++c) {
            int cell = c * NTHREADS + tid;          // 1024 cells: 64 k2 x 16 d4
            int d4   = (cell & 15) << 2;            // column group (4 cols)
            int k2   = cell >> 4;                   // key pair index
            int kg   = tk * KTILE + 2 * k2;
            const float4 va = *(const float4*)(vmat + vBase + (size_t)kg * D_DIM + d4);
            const float4 vb = *(const float4*)(vmat + vBase + (size_t)(kg + 1) * D_DIM + d4);
            VT32[(d4 + 0) * (VT_PAD / 2) + k2] = pack_bf16(va.x, vb.x);
            VT32[(d4 + 1) * (VT_PAD / 2) + k2] = pack_bf16(va.y, vb.y);
            VT32[(d4 + 2) * (VT_PAD / 2) + k2] = pack_bf16(va.z, vb.z);
            VT32[(d4 + 3) * (VT_PAD / 2) + k2] = pack_bf16(va.w, vb.w);
        }
        __syncthreads();

        // ---- 4 x 32-key WMMA passes over the staged tile -------------------
        #pragma unroll
        for (int sub = 0; sub < 4; ++sub) {
            const int kcBase = tk * KTILE + sub * KCHUNK;
            const float* sc = srow + kcBase;

            float4 a0 = *(const float4*)(sc + g0);
            float4 a1 = *(const float4*)(sc + g0 + 4);
            float4 a2 = *(const float4*)(sc + g0 + 16);
            float4 a3 = *(const float4*)(sc + g0 + 20);
            if (tk + 1 < S_DIM / KTILE)
                __builtin_prefetch(sc + KTILE + g0, 0, 3);   // global_prefetch

            float s[16] = { a0.x, a0.y, a0.z, a0.w, a1.x, a1.y, a1.z, a1.w,
                            a2.x, a2.y, a2.z, a2.w, a3.x, a3.y, a3.z, a3.w };

            // ---- online softmax (log2 domain), row = lane&15 ---------------
            float t0 = fmaxf(fmaxf(s[0], s[1]),  fmaxf(s[2], s[3]));
            float t1 = fmaxf(fmaxf(s[4], s[5]),  fmaxf(s[6], s[7]));
            float t2 = fmaxf(fmaxf(s[8], s[9]),  fmaxf(s[10], s[11]));
            float t3 = fmaxf(fmaxf(s[12], s[13]), fmaxf(s[14], s[15]));
            float lm = fmaxf(fmaxf(t0, t1), fmaxf(t2, t3));
            float pm2 = fmaxf(lm, __shfl_xor(lm, 16)) * LOG2E;  // chunk row max
            float m2new = fmaxf(m2, pm2);

            // exp with the 1/keep_p dropout scale pre-baked into the bias:
            //   p'[j] = exp(s - m) / keep_p
            const float bias = LOG2_INV_KEEP - m2new;
            float p[16];
            #pragma unroll
            for (int j = 0; j < 16; ++j)
                p[j] = exp2f(fmaf(s[j], LOG2E, bias));          // v_fma + v_exp

            float ls = ((p[0] + p[1]) + (p[2] + p[3])) + ((p[4] + p[5]) + (p[6] + p[7]))
                     + ((p[8] + p[9]) + (p[10] + p[11])) + ((p[12] + p[13]) + (p[14] + p[15]));
            // true (pre-dropout, unscaled) row sum: undo the 1/keep_p factor
            float lchunk = (ls + __shfl_xor(ls, 16)) * KEEP_P;

            // ---- rescale only when some row's max actually moved -----------
            if (__any(pm2 > m2)) {
                float corr = exp2f(m2 - m2new);
                #pragma unroll
                for (int r = 0; r < 8; ++r) {
                    float cr = __shfl(corr, r + half);
                    #pragma unroll
                    for (int t = 0; t < 4; ++t) acc[t][r] *= cr;
                }
                l = fmaf(l, corr, lchunk);
                m2 = m2new;
            } else {
                l += lchunk;
            }

            // ---- dropout (byte hash, 4 decisions/hash) + pack A ------------
            v8u av;
            #pragma unroll
            for (int b = 0; b < 4; ++b) {
                int klocal = g0 + ((b & 1) << 2) + ((b >> 1) << 4); // 0,4,16,20 (+g0)
                unsigned h = hash4(rowBase + (unsigned)(kcBase + klocal));
                float d0 = ((h       & 255u) < KEEP_BYTE) ? p[b*4+0] : 0.0f;
                float d1 = (((h>>8)  & 255u) < KEEP_BYTE) ? p[b*4+1] : 0.0f;
                float d2 = (((h>>16) & 255u) < KEEP_BYTE) ? p[b*4+2] : 0.0f;
                float d3 = ( (h>>24)         < KEEP_BYTE) ? p[b*4+3] : 0.0f;
                av[2*b]     = pack_bf16(d0, d1);
                av[2*b + 1] = pack_bf16(d2, d3);
            }
            v16bf afrag = __builtin_bit_cast(v16bf, av);

            // ---- B fragments from LDS + WMMA -------------------------------
            #pragma unroll
            for (int t = 0; t < 4; ++t) {
                const uint4* bp = (const uint4*)
                    &VT[(t * 16 + nB) * VT_PAD + sub * KCHUNK + kB]; // 16B aligned
                uint4 b0 = bp[0];                                    // ds_load_b128
                uint4 b1 = bp[2];
                v8u bv;
                bv[0] = b0.x; bv[1] = b0.y; bv[2] = b0.z; bv[3] = b0.w;
                bv[4] = b1.x; bv[5] = b1.y; bv[6] = b1.z; bv[7] = b1.w;
                v16bf bfrag = __builtin_bit_cast(v16bf, bv);

                acc[t] = __builtin_amdgcn_wmma_f32_16x16x32_bf16(
                    false, afrag, false, bfrag, (short)0, acc[t], false, false);
            }
        }
    }

    // ---- epilogue: divide by softmax denominator, store -------------------
    #pragma unroll
    for (int r = 0; r < 8; ++r) {
        float linv = 1.0f / __shfl(l, r + half);
        int qo = qT * BLOCK_Q + wave * 16 + r + half;       // C-space row
        size_t ob = ((size_t)bh * S_DIM + (size_t)qo) * D_DIM + nB;
        #pragma unroll
        for (int t = 0; t < 4; ++t)
            out[ob + t * 16] = acc[t][r] * linv;
    }
}

extern "C" void kernel_launch(void* const* d_in, const int* in_sizes, int n_in,
                              void* d_out, int out_size, void* d_ws, size_t ws_size,
                              hipStream_t stream) {
    (void)in_sizes; (void)n_in; (void)d_ws; (void)ws_size; (void)out_size;
    const float* scores = (const float*)d_in[0];   // [4,16,2048,2048] f32
    const float* vmat   = (const float*)d_in[1];   // [4,16,2048,64]   f32
    float*       out    = (float*)d_out;           // [4,16,2048,64]   f32

    dim3 grid(S_DIM / BLOCK_Q, 4 * 16);            // (8, 64)
    dim3 block(NTHREADS);                           // 16 waves (wave32)
    fused_softmax_dropout_pv<<<grid, block, 0, stream>>>(scores, vmat, out);
}
